// PixContrastive_41678362640818
// MI455X (gfx1250) — matrix operand order repaired and checked by reading, not compiled, not run on hardware
//
#include <hip/hip_runtime.h>
#include <hip/hip_fp16.h>

typedef _Float16 h8  __attribute__((ext_vector_type(8)));
typedef _Float16 h16 __attribute__((ext_vector_type(16)));
typedef float    f8  __attribute__((ext_vector_type(8)));

static constexpr int kN = 8;      // batch
static constexpr int kC = 64;     // channels (= GEMM K)
static constexpr int kP = 4096;   // pixels per image (64*64)

// LDS staging: one stage = all 64 K-rows x 64 q-columns of B (f16).
// Row stride padded to 72 halfs (144 B) to spread bank usage.
static constexpr int kQStage   = 64;                 // q columns per stage
static constexpr int kLdsRow   = 72;                 // halfs per K-row in LDS
static constexpr int kStages   = kP / kQStage;       // 64 (even)

// ---------------------------------------------------------------------------
// Kernel 1: per-pixel L2 normalization over channels + pack to f16 layouts.
//   A-pack (rgb): [n][p][c]  (rows of A, c contiguous)
//   B-pack (ir):  [n][c][q]  (K-major, q contiguous)
// ---------------------------------------------------------------------------
__global__ void __launch_bounds__(256)
norm_pack(const float* __restrict__ rgb, const float* __restrict__ ir,
          _Float16* __restrict__ Apack, _Float16* __restrict__ Bpack) {
  int idx = blockIdx.x * blockDim.x + threadIdx.x;   // n*kP + p
  if (idx >= kN * kP) return;
  int n = idx >> 12;
  int p = idx & (kP - 1);

  const float* a = rgb + (size_t)n * kC * kP + p;
  const float* b = ir  + (size_t)n * kC * kP + p;

  float sa = 0.f, sb = 0.f;
#pragma unroll 8
  for (int c = 0; c < kC; ++c) { float v = a[(size_t)c * kP]; sa += v * v; }
#pragma unroll 8
  for (int c = 0; c < kC; ++c) { float v = b[(size_t)c * kP]; sb += v * v; }
  float ra = 1.0f / fmaxf(sqrtf(sa), 1e-12f);
  float rb = 1.0f / fmaxf(sqrtf(sb), 1e-12f);

  _Float16* arow = Apack + ((size_t)n * kP + p) * kC;    // [n][p][c]
  _Float16* bcol = Bpack + (size_t)n * kC * kP + p;      // [n][c][q]
#pragma unroll 8
  for (int c = 0; c < kC; ++c) arow[c] = (_Float16)(a[(size_t)c * kP] * ra);
#pragma unroll 8
  for (int c = 0; c < kC; ++c) bcol[(size_t)c * kP] = (_Float16)(b[(size_t)c * kP] * rb);
}

// ---------------------------------------------------------------------------
// Kernel 2: tiled S = A*B with B staged in LDS via async global->LDS DMA,
// double-buffered; exp(S/T) fused reduction (tot + diagonal).
// Block = 8 waves = 8 consecutive p-tiles of one batch; all waves share the
// same q sweep, so one LDS stage serves the whole block (8x L2 traffic cut).
// ---------------------------------------------------------------------------
__global__ void __launch_bounds__(256)
gemm_exp_reduce(const _Float16* __restrict__ Apack,
                const _Float16* __restrict__ Bpack,
                float* __restrict__ acc) {
  __shared__ _Float16 bufLds[2][kC * kLdsRow];   // 2 x 9 KB double buffer

  const int lane   = threadIdx.x & 31;
  const int waveId = threadIdx.x >> 5;
  const int n      = blockIdx.x >> 5;            // 32 blocks per batch
  const int blkInB = blockIdx.x & 31;
  // p0 is wave-uniform; readfirstlane makes it *provably* scalar so the
  // diagonal test compiles to s_cmp + uniform branch (not per-lane cndmask),
  // and the A-fragment base becomes an SGPR.
  const int p0 = __builtin_amdgcn_readfirstlane(((blkInB << 3) + waveId) << 4);

  const int g = lane >> 4;          // lane group (0/1)
  const int r = lane & 15;          // row within tile

  // --- A fragments (16-bit A 16x32 layout): lane (g,r) holds row p0+r,
  //     K = kb + g*8 + [0..7] and K = kb + 16 + g*8 + [0..7].
  const _Float16* arow = Apack + ((size_t)n * kP + (p0 + r)) * kC;
  h8 a0lo = *(const h8*)(arow + 0  + g * 8);
  h8 a0hi = *(const h8*)(arow + 16 + g * 8);
  h8 a1lo = *(const h8*)(arow + 32 + g * 8);
  h8 a1hi = *(const h8*)(arow + 48 + g * 8);
  h16 afrag0 = __builtin_shufflevector(a0lo, a0hi, 0,1,2,3,4,5,6,7,8,9,10,11,12,13,14,15);
  h16 afrag1 = __builtin_shufflevector(a1lo, a1hi, 0,1,2,3,4,5,6,7,8,9,10,11,12,13,14,15);

  // --- Async staging map. One stage = kC rows x kQStage cols = 512 chunks of
  //     16 B. Each wave issues 2 global_load_async_to_lds_b128 per stage:
  //     chunk t = (waveId*2 + j)*32 + lane;  K-row k = t/4, chunk-in-row h = t%4.
  const _Float16* bB = Bpack + (size_t)n * kC * kP;   // batch B base (SGPR64)
  unsigned lds0[2], lds1[2], gOffB[2];
#pragma unroll
  for (int j = 0; j < 2; ++j) {
    int t = ((waveId * 2 + j) << 5) + lane;           // 0..511
    int k = t >> 2, h = t & 3;
    gOffB[j] = (unsigned)(k * kP + h * 8) * 2u;       // bytes within batch B
    lds0[j]  = (unsigned)(uintptr_t)&bufLds[0][k * kLdsRow + h * 8];
    lds1[j]  = (unsigned)(uintptr_t)&bufLds[1][k * kLdsRow + h * 8];
  }

  // Issue one stage's DMA into the given buffer (by explicit address array,
  // so no per-iteration buffer-parity selects).
  auto issueInto = [&](const unsigned* lds, int stage) {
    const unsigned qByte = (unsigned)(stage * kQStage) * 2u;
#pragma unroll
    for (int j = 0; j < 2; ++j) {
      unsigned go = gOffB[j] + qByte;
      // GVS form: LDS-addr VGPR, 32-bit offset VGPR, SGPR64 base. ASYNCcnt.
      asm volatile("global_load_async_to_lds_b128 %0, %1, %2"
                   :: "v"(lds[j]), "v"(go), "s"(bB)
                   : "memory");
    }
  };

  const float SCALE = 10.0f * 1.44269504088896340736f;  // (1/T) * log2(e)
  // C/D layout: M = v + 8*(lane/16), N = lane%16; diagonal <=> v == dv.
  const int dv = r - (g << 3);
  float sumTot = 0.f, sumDiag = 0.f;

  // Compute all 4 sub-tiles of a staged buffer.
  auto computeStage = [&](const _Float16* bufc, int stage) {
    const int qb = stage * kQStage;
#pragma unroll
    for (int s = 0; s < kQStage / 16; ++s) {
      // B fragments (32x16 f16): lane holds column s*16+r, K = kb + g*8 + [0..7] and K = kb + 16 + g*8 + [0..7].
      const _Float16* c0 = bufc + (0  + g * 8) * kLdsRow + s * 16 + r;  // kb = 0
      const _Float16* c1 = bufc + (32 + g * 8) * kLdsRow + s * 16 + r;  // kb = 32
      h16 bfrag0, bfrag1;
#pragma unroll
      for (int i = 0; i < 8; ++i) {
        bfrag0[i] = c0[i * kLdsRow];
        bfrag0[i + 8] = c0[(16 + i) * kLdsRow];
        bfrag1[i] = c1[i * kLdsRow];
        bfrag1[i + 8] = c1[(16 + i) * kLdsRow];
      }

      f8 cacc = {};
      cacc = __builtin_amdgcn_wmma_f32_16x16x32_f16(
          false, afrag0, false, bfrag0, (short)0, cacc, false, false);
      cacc = __builtin_amdgcn_wmma_f32_16x16x32_f16(
          false, afrag1, false, bfrag1, (short)0, cacc, false, false);

      float e[8];
#pragma unroll
      for (int i = 0; i < 8; ++i) {
        e[i] = __builtin_amdgcn_exp2f(cacc[i] * SCALE);
        sumTot += e[i];
      }
      // Uniform (scalar) comparison -> s_cbranch; taken once per sweep.
      if (qb + s * 16 == p0) {
#pragma unroll
        for (int i = 0; i < 8; ++i) sumDiag += (dv == i) ? e[i] : 0.f;
      }
    }
  };

  // Prologue: stage 0 in flight into buf0, then wait + barrier before use.
  issueInto(lds0, 0);
  asm volatile("s_wait_asynccnt 0x0" ::: "memory");
  __syncthreads();

  // Double-buffered pipeline, unrolled by 2 so buffer choice is static.
  for (int stage = 0; stage < kStages; stage += 2) {
    // even stage: compute buf0, DMA next into buf1
    if (stage + 1 < kStages) issueInto(lds1, stage + 1);
    computeStage(&bufLds[0][0], stage);
    asm volatile("s_wait_asynccnt 0x0" ::: "memory");
    __syncthreads();
    // odd stage: compute buf1, DMA next into buf0
    if (stage + 2 < kStages) issueInto(lds0, stage + 2);
    computeStage(&bufLds[1][0], stage + 1);
    asm volatile("s_wait_asynccnt 0x0" ::: "memory");
    __syncthreads();
  }

  // wave32 tree reduction
#pragma unroll
  for (int off = 16; off > 0; off >>= 1) {
    sumTot  += __shfl_xor(sumTot,  off, 32);
    sumDiag += __shfl_xor(sumDiag, off, 32);
  }
  if (lane == 0) {
    atomicAdd(&acc[2 * n + 0], sumTot);
    atomicAdd(&acc[2 * n + 1], sumDiag);
  }
}

// ---------------------------------------------------------------------------
// Init + finalize
// ---------------------------------------------------------------------------
__global__ void zero_acc(float* __restrict__ acc) {
  if (threadIdx.x < 2 * kN) acc[threadIdx.x] = 0.f;
}

__global__ void finalize(const float* __restrict__ acc, float* __restrict__ out) {
  if (threadIdx.x == 0 && blockIdx.x == 0) {
    float loss = 0.f;
#pragma unroll
    for (int n = 0; n < kN; ++n) {
      float tot = acc[2 * n + 0];
      float pos = acc[2 * n + 1];
      loss += -logf(pos / (tot + 1e-6f));
    }
    out[0] = loss / (float)kN;
  }
}

// ---------------------------------------------------------------------------
extern "C" void kernel_launch(void* const* d_in, const int* in_sizes, int n_in,
                              void* d_out, int out_size, void* d_ws, size_t ws_size,
                              hipStream_t stream) {
  const float* rgb = (const float*)d_in[0];
  const float* ir  = (const float*)d_in[1];
  // d_in[2] (targets) is unused by the forward math.

  char* ws = (char*)d_ws;
  const size_t packBytes = (size_t)kN * kP * kC * sizeof(_Float16);  // 4 MB each
  _Float16* Apack = (_Float16*)ws;
  _Float16* Bpack = (_Float16*)(ws + packBytes);
  float*    acc   = (float*)(ws + 2 * packBytes);

  zero_acc<<<1, 32, 0, stream>>>(acc);
  norm_pack<<<(kN * kP) / 256, 256, 0, stream>>>(rgb, ir, Apack, Bpack);
  // 2048 waves total, 8 waves (256 threads) per block -> 256 blocks.
  gemm_exp_reduce<<<256, 256, 0, stream>>>(Apack, Bpack, acc);
  finalize<<<1, 32, 0, stream>>>(acc, (float*)d_out);
}
